// ProductKeyAttention_61821759259200
// MI455X (gfx1250) — matrix operand order repaired and checked by reading, not compiled, not run on hardware
//
#include <hip/hip_runtime.h>

// ---------------------------------------------------------------------------
// ProductKeyAttention for MI455X (gfx1250, wave32, WMMA).
// Pipeline: q-GEMM (bf16 WMMA) -> pq-GEMM -> PK top-k routing -> weighted
// embedding-bag gather -> flash causal attention (bf16 WMMA, double-buffered
// async-to-LDS k/v staging) -> out-GEMM.
// ---------------------------------------------------------------------------

typedef __attribute__((ext_vector_type(16))) __bf16 v16bf;
typedef __attribute__((ext_vector_type(8)))  float  v8f;
typedef int v4i __attribute__((vector_size(4 * sizeof(int))));

// Model dims (fixed by the reference)
#define BB   2
#define SS   1024
#define DD   1024
#define HH   8
#define NKV  1024
#define NK   32
#define DK   64
#define TK   8

// ---- async global->LDS copy (GLOBAL_LOAD_ASYNC_TO_LDS_B128, ASYNCcnt) ----
#if defined(__has_builtin)
# if __has_builtin(__builtin_amdgcn_global_load_async_to_lds_b128)
#  define HAVE_ASYNC_LDS 1
# endif
#endif
#ifndef HAVE_ASYNC_LDS
# define HAVE_ASYNC_LDS 0
#endif

typedef v4i __attribute__((address_space(1)))* gv4i_p;   // global int4*
typedef v4i __attribute__((address_space(3)))* lv4i_p;   // LDS int4*

__device__ __forceinline__ void cp16(void* l, const void* g) {
#if HAVE_ASYNC_LDS
  __builtin_amdgcn_global_load_async_to_lds_b128(
      (gv4i_p)(v4i*)const_cast<void*>(g),
      (lv4i_p)(v4i*)l,
      0, 0);
#else
  *(uint4*)l = *(const uint4*)g;
#endif
}

__device__ __forceinline__ void wait_async0() {
#if HAVE_ASYNC_LDS
# if __has_builtin(__builtin_amdgcn_s_wait_asynccnt)
  __builtin_amdgcn_s_wait_asynccnt(0);
# else
  asm volatile("s_wait_asynccnt 0" ::: "memory");
# endif
#endif
}

__device__ __forceinline__ __bf16 f2bf(float f) { return (__bf16)f; }  // native v_cvt, RNE

// 16-bit A-matrix (16x32) per-lane K index (ISA 7.12.2):
// lanes 0-15: K = 0..7 then 16..23 ; lanes 16-31: K = 8..15 then 24..31
__device__ __forceinline__ int a_kidx(int lane, int j) {
  int base = (lane >> 4) * 8;
  return (j < 8) ? (base + j) : (16 + base + (j - 8));
}

// ---------------------------------------------------------------------------
// Tiled GEMM:  C[M,N] = scale * A[M,K] (f32) x W[N,K]^T (f32), bf16 WMMA core.
// mode 0: store f32 row-major [M,N].
// mode 1: store bf16 permuted [B,H,S,D] (m = b*S+s, n = h*D+d)  -- for q.
// Block = 256 threads = 8 waves; block tile 128x128; wave tile 32x64.
// ---------------------------------------------------------------------------
__global__ __launch_bounds__(256) void gemm_wmma_bf16(
    const float* __restrict__ A, const float* __restrict__ W,
    float* __restrict__ outF, __bf16* __restrict__ outB,
    int M, int N, int K, float scale, int mode)
{
  __shared__ __bf16 ldsA[128 * 32];
  __shared__ __bf16 ldsB[128 * 32];

  const int tid  = threadIdx.x;
  const int lane = tid & 31;
  const int w    = tid >> 5;
  const int mw   = w & 3;            // 4 M-waves
  const int nw   = w >> 2;           // 2 N-waves
  const int mBase = blockIdx.y * 128;
  const int nBase = blockIdx.x * 128;
  const int mo = mw * 32, no = nw * 64;

  v8f acc[2][4];
#pragma unroll
  for (int i = 0; i < 2; ++i)
#pragma unroll
    for (int t = 0; t < 4; ++t)
#pragma unroll
      for (int r = 0; r < 8; ++r) acc[i][t][r] = 0.0f;

  const int lr = tid >> 1;           // 0..127 tile row to load
  const int lh = (tid & 1) * 16;     // k-half

  for (int kt = 0; kt < K; kt += 32) {
    const float* aRow = A + (size_t)(mBase + lr) * K + kt + lh;
    const float* bRow = W + (size_t)(nBase + lr) * K + kt + lh;
#pragma unroll
    for (int j = 0; j < 16; ++j) {
      ldsA[lr * 32 + lh + j] = f2bf(aRow[j]);
      ldsB[lr * 32 + lh + j] = f2bf(bRow[j]);
    }
    __syncthreads();

    v16bf a[2], b[4];
#pragma unroll
    for (int i = 0; i < 2; ++i)
#pragma unroll
      for (int j = 0; j < 16; ++j)
        a[i][j] = ldsA[(mo + i * 16 + (lane & 15)) * 32 + (lane >> 4) * 16 + j];
#pragma unroll
    for (int t = 0; t < 4; ++t)
#pragma unroll
      for (int j = 0; j < 16; ++j)
        b[t][j] = ldsB[(no + t * 16 + (lane & 15)) * 32 + (lane >> 4) * 16 + j];

#pragma unroll
    for (int i = 0; i < 2; ++i)
#pragma unroll
      for (int t = 0; t < 4; ++t)
        acc[i][t] = __builtin_amdgcn_wmma_f32_16x16x32_bf16(
            false, a[i], false, b[t], (short)0, acc[i][t], false, false);
    __syncthreads();
  }

#pragma unroll
  for (int i = 0; i < 2; ++i)
#pragma unroll
    for (int t = 0; t < 4; ++t)
#pragma unroll
      for (int r = 0; r < 8; ++r) {
        int m = mBase + mo + i * 16 + r + ((lane >> 4) << 3);
        int n = nBase + no + t * 16 + (lane & 15);
        float val = acc[i][t][r] * scale;
        if (mode == 0) {
          outF[(size_t)m * N + n] = val;
        } else {
          int b_ = m / SS, s_ = m % SS, h_ = n / DD, d_ = n % DD;
          outB[(((size_t)b_ * HH + h_) * SS + s_) * DD + d_] = f2bf(val);
        }
      }
}

// ---------------------------------------------------------------------------
// Product-key routing: one wave per (b*s, h). Lane n = sub-key n (NK=32).
// Per sub-space top-8 via wave argmax, 8x8 combine, top-8 of 64, softmax.
// ---------------------------------------------------------------------------
__global__ __launch_bounds__(256) void pk_route_kernel(
    const float* __restrict__ pq, const float* __restrict__ pk_keys,
    int* __restrict__ outIdx, float* __restrict__ outW)
{
  const int lane = threadIdx.x & 31;
  const int w    = threadIdx.x >> 5;
  const int gw   = blockIdx.x * 8 + w;     // (bs, h)
  const int h    = gw & (HH - 1);
  const int bs   = gw >> 3;

  float sim[2];
#pragma unroll
  for (int p = 0; p < 2; ++p) {
    const float* qv = pq + (size_t)bs * (2 * HH * DK) + (p * HH + h) * DK;
    const float* kv = pk_keys + (size_t)((p * NK + lane) * HH + h) * DK;
    float acc = 0.f;
    for (int d = 0; d < DK; ++d) acc += qv[d] * kv[d];
    sim[p] = acc;
  }

  float s0[8], s1[8]; int i0[8], i1[8];
  float v0 = sim[0], v1 = sim[1];
#pragma unroll
  for (int t = 0; t < 8; ++t) {
    float bv = v0; int bi = lane;
    for (int off = 16; off; off >>= 1) {
      float ov = __shfl_xor(bv, off, 32);
      int   oi = __shfl_xor(bi, off, 32);
      if (ov > bv || (ov == bv && oi < bi)) { bv = ov; bi = oi; }
    }
    s0[t] = bv; i0[t] = bi;
    if (lane == bi) v0 = -1e30f;

    float cv = v1; int ci = lane;
    for (int off = 16; off; off >>= 1) {
      float ov = __shfl_xor(cv, off, 32);
      int   oi = __shfl_xor(ci, off, 32);
      if (ov > cv || (ov == cv && oi < ci)) { cv = ov; ci = oi; }
    }
    s1[t] = cv; i1[t] = ci;
    if (lane == ci) v1 = -1e30f;
  }

  // Combined 64 candidates: lane holds combos (lane) and (lane+32).
  float ca = s0[lane >> 3]       + s1[lane & 7];
  float cb = s0[4 + (lane >> 3)] + s1[lane & 7];

  float fsc[8]; int fid[8];
#pragma unroll
  for (int t = 0; t < 8; ++t) {
    float lv; int lc;
    if (ca >= cb) { lv = ca; lc = lane; } else { lv = cb; lc = lane + 32; }
    for (int off = 16; off; off >>= 1) {
      float ov = __shfl_xor(lv, off, 32);
      int   oc = __shfl_xor(lc, off, 32);
      if (ov > lv || (ov == lv && oc < lc)) { lv = ov; lc = oc; }
    }
    fsc[t] = lv;
    int t0 = lc >> 3, t1 = lc & 7;
    fid[t] = i0[t0] + i1[t1] * NK;
    if (lane == (lc & 31)) { if (lc < 32) ca = -1e30f; else cb = -1e30f; }
  }

  float mx = fsc[0];
  for (int t = 1; t < 8; ++t) mx = fmaxf(mx, fsc[t]);
  float e[8], sum = 0.f;
  for (int t = 0; t < 8; ++t) { e[t] = __expf(fsc[t] - mx); sum += e[t]; }
  if (lane == 0) {
    for (int t = 0; t < 8; ++t) {
      outIdx[((size_t)bs * HH + h) * TK + t] = fid[t];
      outW  [((size_t)bs * HH + h) * TK + t] = e[t] / sum;
    }
  }
}

// ---------------------------------------------------------------------------
// Weighted embedding-bag gather -> bf16 k,v in [B,H,S,D].
// One 256-thread block per (b,s,h); float4 coalesced row reads.
// ---------------------------------------------------------------------------
__global__ __launch_bounds__(256) void pk_gather_kernel(
    const float* __restrict__ keys_emb, const float* __restrict__ values_emb,
    const int* __restrict__ idxBuf, const float* __restrict__ wBuf,
    __bf16* __restrict__ kOut, __bf16* __restrict__ vOut)
{
  const int gid = blockIdx.x;            // bs*H + h
  const int h   = gid & (HH - 1);
  const int bs  = gid >> 3;
  const int b   = bs / SS, s = bs % SS;
  const int d0  = threadIdx.x * 4;       // 1024 / 256

  int   idx[TK];
  float wgt[TK];
#pragma unroll
  for (int t = 0; t < TK; ++t) {
    idx[t] = idxBuf[(size_t)gid * TK + t] + h * NKV;
    wgt[t] = wBuf [(size_t)gid * TK + t];
  }

  float ak[4] = {0,0,0,0}, av[4] = {0,0,0,0};
#pragma unroll
  for (int t = 0; t < TK; ++t) {
    const float4 kr = *(const float4*)(keys_emb   + (size_t)idx[t] * DD + d0);
    const float4 vr = *(const float4*)(values_emb + (size_t)idx[t] * DD + d0);
    ak[0] += wgt[t] * kr.x; ak[1] += wgt[t] * kr.y;
    ak[2] += wgt[t] * kr.z; ak[3] += wgt[t] * kr.w;
    av[0] += wgt[t] * vr.x; av[1] += wgt[t] * vr.y;
    av[2] += wgt[t] * vr.z; av[3] += wgt[t] * vr.w;
  }
  const size_t off = (((size_t)b * HH + h) * SS + s) * DD + d0;
#pragma unroll
  for (int j = 0; j < 4; ++j) { kOut[off + j] = f2bf(ak[j]); vOut[off + j] = f2bf(av[j]); }
}

// ---------------------------------------------------------------------------
// Flash-style causal attention, d_head = 1024.
// Block = 256 threads (8 waves) handles 16 query rows of one (b,h).
// Key tiles of 32, DOUBLE-BUFFERED in LDS: tile t+1 is prefetched with
// GLOBAL_LOAD_ASYNC_TO_LDS_B128 (ASYNCcnt) while tile t is computed.
// Scores: each wave reduces a 128-wide K-slice (4 WMMA ksteps x 2 n-tiles),
// cross-wave LDS reduce. PV: wave owns 128 out columns (8 WMMA, K=32).
// LDS: q 32KB + 2x(k 64KB + v 64KB) + reduce/softmax ~19.6KB = 314,560 B
// (< 320KB WGP limit; 1 WG/WGP by design -- kernel is WMMA/LDS dense).
// ---------------------------------------------------------------------------
#define ATTN_LDS_BYTES (32768 + 2 * 65536 + 2 * 65536 + 16384 + 2048 + 192 + 1024)

__global__ __launch_bounds__(256) void attn_kernel(
    const __bf16* __restrict__ qbf, const __bf16* __restrict__ kbf,
    const __bf16* __restrict__ vbf, float* __restrict__ outAcc)
{
  extern __shared__ char smem[];
  __bf16* qT    = (__bf16*)smem;                    // 16 x 1024
  __bf16* kT[2] = { qT + 16 * 1024, qT + 16 * 1024 + 32 * 1024 };
  __bf16* vT[2] = { kT[1] + 32 * 1024, kT[1] + 32 * 1024 + 32 * 1024 };
  float*  sPart = (float*)(vT[1] + 32 * 1024);      // 8 x 16 x 32
  float*  sS    = sPart + 8 * 16 * 32;              // 16 x 32
  float*  sRowM = sS + 16 * 32;                     // 16
  float*  sRowL = sRowM + 16;                       // 16
  float*  sCorr = sRowL + 16;                       // 16
  __bf16* sP    = (__bf16*)(sCorr + 16);            // 16 x 32

  const int tid  = threadIdx.x;
  const int lane = tid & 31;
  const int w    = tid >> 5;
  const int qt   = blockIdx.x & 63;                 // S/16
  const int h    = (blockIdx.x >> 6) & 7;
  const int b    = blockIdx.x >> 9;
  const int q0   = qt * 16;
  const size_t headOff = (((size_t)b * HH + h) * SS) * DD;

  // ---- issue async staging for q tile (32KB) and k/v tile 0 (64KB each) ----
  {
    const char* qsrc = (const char*)(qbf + headOff + (size_t)q0 * DD);
    char* qdst = (char*)qT;
    for (int i = tid; i < 2048; i += 256) cp16(qdst + i * 16, qsrc + i * 16);
    const char* ksrc = (const char*)(kbf + headOff);
    const char* vsrc = (const char*)(vbf + headOff);
    char* kdst = (char*)kT[0];
    char* vdst = (char*)vT[0];
    for (int i = tid; i < 4096; i += 256) {
      cp16(kdst + i * 16, ksrc + i * 16);
      cp16(vdst + i * 16, vsrc + i * 16);
    }
  }
  if (tid < 16) { sRowM[tid] = -1e30f; sRowL[tid] = 0.f; }

  v8f acc[8];
#pragma unroll
  for (int t = 0; t < 8; ++t)
#pragma unroll
    for (int r = 0; r < 8; ++r) acc[t][r] = 0.0f;

  const int nKt = (q0 + 15) / 32 + 1;
  for (int kt = 0; kt < nKt; ++kt) {
    const int k0  = kt * 32;
    const int cur = kt & 1;
    wait_async0();                 // tile kt (and q on iter 0) resident in LDS
    __syncthreads();               // cross-wave visibility + buffer reuse safety

    // ---- prefetch tile kt+1 into the other buffer while computing kt ----
    if (kt + 1 < nKt) {
      const char* ksrc = (const char*)(kbf + headOff + (size_t)(k0 + 32) * DD);
      const char* vsrc = (const char*)(vbf + headOff + (size_t)(k0 + 32) * DD);
      char* kdst = (char*)kT[cur ^ 1];
      char* vdst = (char*)vT[cur ^ 1];
      for (int i = tid; i < 4096; i += 256) {
        cp16(kdst + i * 16, ksrc + i * 16);
        cp16(vdst + i * 16, vsrc + i * 16);
      }
    }
    const __bf16* kC = kT[cur];
    const __bf16* vC = vT[cur];

    // ---- partial scores: wave w reduces d in [w*128, w*128+128) ----
    v8f c0, c1;
#pragma unroll
    for (int r = 0; r < 8; ++r) { c0[r] = 0.f; c1[r] = 0.f; }
    const int ds0 = w * 128;
#pragma unroll
    for (int ks = 0; ks < 4; ++ks) {
      const int koff = ds0 + ks * 32;
      v16bf a, b0, b1;
#pragma unroll
      for (int j = 0; j < 16; ++j) {
        a [j] = qT[(lane & 15) * 1024 + koff + (lane >> 4) * 16 + j];
        b0[j] = kC[(lane & 15) * 1024 + koff + (lane >> 4) * 16 + j];
        b1[j] = kC[((lane & 15) + 16) * 1024 + koff + (lane >> 4) * 16 + j];
      }
      c0 = __builtin_amdgcn_wmma_f32_16x16x32_bf16(false, a, false, b0, (short)0, c0, false, false);
      c1 = __builtin_amdgcn_wmma_f32_16x16x32_bf16(false, a, false, b1, (short)0, c1, false, false);
    }
#pragma unroll
    for (int r = 0; r < 8; ++r) {
      int m = r + ((lane >> 4) << 3);
      sPart[(w * 16 + m) * 32 + (lane & 15)]      = c0[r];
      sPart[(w * 16 + m) * 32 + 16 + (lane & 15)] = c1[r];
    }
    __syncthreads();

    // ---- cross-wave reduce + causal mask ----
    for (int e = tid; e < 512; e += 256) {
      int m = e >> 5, n = e & 31;
      float s = 0.f;
#pragma unroll
      for (int ww = 0; ww < 8; ++ww) s += sPart[(ww * 16 + m) * 32 + n];
      if (k0 + n > q0 + m) s = -1e30f;
      sS[m * 32 + n] = s;
    }
    __syncthreads();

    // ---- online softmax (16 rows by 16 threads) ----
    if (tid < 16) {
      const int m = tid;
      float mx = -1e30f;
      for (int n = 0; n < 32; ++n) mx = fmaxf(mx, sS[m * 32 + n]);
      float mOld = sRowM[m];
      float mNew = fmaxf(mOld, mx);
      float corr = __expf(mOld - mNew);
      float sum = 0.f;
      for (int n = 0; n < 32; ++n) {
        float p = __expf(sS[m * 32 + n] - mNew);
        sP[m * 32 + n] = f2bf(p);
        sum += p;
      }
      sRowL[m] = sRowL[m] * corr + sum;
      sRowM[m] = mNew;
      sCorr[m] = corr;
    }
    __syncthreads();

    // ---- rescale accumulators + P x V (wave owns cols [w*128, w*128+128)) ----
    v16bf ap;
#pragma unroll
    for (int j = 0; j < 16; ++j)
      ap[j] = sP[(lane & 15) * 32 + (lane >> 4) * 16 + j];
#pragma unroll
    for (int t = 0; t < 8; ++t) {
#pragma unroll
      for (int r = 0; r < 8; ++r) acc[t][r] *= sCorr[r + ((lane >> 4) << 3)];
      v16bf bv;
      const int col = w * 128 + t * 16 + (lane & 15);
#pragma unroll
      for (int j = 0; j < 16; ++j)
        bv[j] = vC[((lane >> 4) * 16 + j) * 1024 + col];
      acc[t] = __builtin_amdgcn_wmma_f32_16x16x32_bf16(false, ap, false, bv, (short)0, acc[t], false, false);
    }
  }

  // ---- normalize and store to [B,S,H*D] for the output GEMM ----
#pragma unroll
  for (int t = 0; t < 8; ++t)
#pragma unroll
    for (int r = 0; r < 8; ++r) {
      int m = r + ((lane >> 4) << 3);
      float l = sRowL[m];
      float val = acc[t][r] / fmaxf(l, 1e-20f);
      size_t off = ((size_t)(b * SS + q0 + m)) * (HH * DD)
                 + (size_t)h * DD + w * 128 + t * 16 + (lane & 15);
      outAcc[off] = val;
    }
}

// ---------------------------------------------------------------------------
// Host-side orchestration. Workspace layout (needs ~169MB):
//   [0,32M)   q   bf16 [B,H,S,D]
//   [32,64M)  k   bf16 [B,H,S,D]
//   [64,96M)  v   bf16 [B,H,S,D]
//   [96,104M) pq  f32  [B*S, 1024]
//   [104M, +512K)       pk indices int32 [B,S,H,TK]
//   [104M+512K, +512K)  pk weights f32
//   [105M,169M) attn out f32 [B*S, H*D]
// ---------------------------------------------------------------------------
extern "C" void kernel_launch(void* const* d_in, const int* in_sizes, int n_in,
                              void* d_out, int out_size, void* d_ws, size_t ws_size,
                              hipStream_t stream) {
  const float* inputs     = (const float*)d_in[0];
  const float* Wq         = (const float*)d_in[1];
  const float* Wpk        = (const float*)d_in[2];
  const float* pk_keys    = (const float*)d_in[3];
  const float* keys_emb   = (const float*)d_in[4];
  const float* values_emb = (const float*)d_in[5];
  const float* Wo         = (const float*)d_in[6];
  float* out = (float*)d_out;

  char* ws = (char*)d_ws;
  __bf16* qbf   = (__bf16*)(ws);
  __bf16* kbf   = (__bf16*)(ws + (32ull << 20));
  __bf16* vbf   = (__bf16*)(ws + (64ull << 20));
  float*  pq    = (float*)(ws + (96ull << 20));
  int*    pkIdx = (int*)  (ws + (104ull << 20));
  float*  pkW   = (float*)(ws + (104ull << 20) + (512u << 10));
  float*  aOut  = (float*)(ws + (105ull << 20));

  const int M = BB * SS;                 // 2048

  // 1) q = scale * x @ Wq^T  -> bf16 [B,H,S,D]   (scale = D^-0.5 = 1/32)
  gemm_wmma_bf16<<<dim3((HH * DD) / 128, M / 128), 256, 0, stream>>>(
      inputs, Wq, nullptr, qbf, M, HH * DD, DD, 0.03125f, 1);

  // 2) pq = x @ Wpk^T -> f32 [2048, 1024]
  gemm_wmma_bf16<<<dim3(1024 / 128, M / 128), 256, 0, stream>>>(
      inputs, Wpk, pq, nullptr, M, 1024, DD, 1.0f, 0);

  // 3) product-key routing (one wave per (b,s,h))
  pk_route_kernel<<<(M * HH) / 8, 256, 0, stream>>>(pq, pk_keys, pkIdx, pkW);

  // 4) weighted embedding-bag gathers -> bf16 k, v
  pk_gather_kernel<<<M * HH, 256, 0, stream>>>(keys_emb, values_emb,
                                               pkIdx, pkW, kbf, vbf);

  // 5) causal flash attention (async double-buffered k/v staging)
  (void)hipFuncSetAttribute((const void*)attn_kernel,
                            hipFuncAttributeMaxDynamicSharedMemorySize,
                            ATTN_LDS_BYTES);
  attn_kernel<<<BB * HH * (SS / 16), 256, ATTN_LDS_BYTES, stream>>>(
      qbf, kbf, vbf, aOut);

  // 6) out = attn @ Wo^T -> d_out f32 [2048, 1024]
  gemm_wmma_bf16<<<dim3(DD / 128, M / 128), 256, 0, stream>>>(
      aOut, Wo, out, nullptr, M, DD, HH * DD, 1.0f, 0);
}
